// GRU_71090298683743
// MI455X (gfx1250) — compile-verified, matrix operand order checked by reading, not run on hardware
//
#include <hip/hip_runtime.h>
#include <hip/hip_bf16.h>

// ---------------------------------------------------------------------------
// GRU-variant recurrence for MI455X (gfx1250, wave32, WMMA).
//   Phase A : bf16 WMMA GEMM for hoisted input projections, double-buffered
//             async global->LDS staging (ASYNCcnt).
//   Phase B : persistent-kernel recurrence, 32 resident WGs, weights pinned
//             in LDS, whole hidden state async-staged to LDS per step.
// ---------------------------------------------------------------------------

typedef unsigned short u16;
typedef __attribute__((ext_vector_type(16))) __bf16 v16bf;
typedef __attribute__((ext_vector_type(8)))  float  v8f;
typedef int v4i __attribute__((vector_size(16)));
static_assert(sizeof(v16bf) == 32, "v16bf must be 8 VGPRs");

#define B_   64
#define T_   2048
#define D_   512
#define H_   512
#define H2_  1024
#define NWG  32

// ---- CDNA5 async global->LDS (guarded: falls back to sync copy) -----------
#if defined(__has_builtin)
#  if __has_builtin(__builtin_amdgcn_global_load_async_to_lds_b128)
#    define HAS_ASYNC_LDS 1
#  endif
#  if __has_builtin(__builtin_amdgcn_s_wait_asynccnt)
#    define HAS_WAIT_ASYNC 1
#  endif
#endif
#ifndef HAS_ASYNC_LDS
#  define HAS_ASYNC_LDS 0
#endif
#ifndef HAS_WAIT_ASYNC
#  define HAS_WAIT_ASYNC 0
#endif

#if HAS_ASYNC_LDS
typedef __attribute__((address_space(1))) v4i* gv4i_t;   // global int4*
typedef __attribute__((address_space(3))) v4i* lv4i_t;   // LDS int4*
#endif

__device__ __forceinline__ void async_copy16(const u16* g, u16* l) {
#if HAS_ASYNC_LDS
  __builtin_amdgcn_global_load_async_to_lds_b128((gv4i_t)g, (lv4i_t)l, 0, 0);
#else
  *(uint4*)l = *(const uint4*)g;
#endif
}

__device__ __forceinline__ void wait_async0() {
#if HAS_ASYNC_LDS
#  if HAS_WAIT_ASYNC
  __builtin_amdgcn_s_wait_asynccnt(0);
#  else
  asm volatile("s_wait_asynccnt 0x0" ::: "memory");
#  endif
#endif
}

// bulk flat copy: elems must be a multiple of 8 (16B chunks), 128 threads
__device__ __forceinline__ void bulk_copy(u16* lds, const u16* g, int elems, int tid) {
  const int n16 = elems >> 3;
  for (int i = tid; i < n16; i += 128)
    async_copy16(g + i * 8, lds + i * 8);
}

// ---- helpers ---------------------------------------------------------------

__device__ __forceinline__ u16 f2bf(float f) {           // round-to-nearest-even
  unsigned u = __float_as_uint(f);
  unsigned r = (u + 0x7FFFu + ((u >> 16) & 1u)) >> 16;
  return (u16)r;
}

union FragU { uint4 q[2]; v16bf v; };

// 16x32 bf16 WMMA operand fragment from row-major [rows, pitch] LDS tile.
//   lanes 0-15 : row r, K = {0..7, 16..23};  lanes 16-31: K = {8..15, 24..31}
__device__ __forceinline__ v16bf load_frag(const u16* base, int row0, int kOff,
                                           int pitch, int lane) {
  int r  = row0 + (lane & 15);
  int kq = (lane >> 4) << 3;
  const u16* p = base + r * pitch + kOff + kq;
  FragU f;
  f.q[0] = *(const uint4*)(p);
  f.q[1] = *(const uint4*)(p + 16);
  return f.v;
}

__device__ __forceinline__ float sigmoidf(float x) {
  return 1.0f / (1.0f + __expf(-x));
}

// Monotonic device-wide barrier (all NWG workgroups resident).
__device__ __forceinline__ void grid_barrier(int* cnt) {
  __syncthreads();
  __threadfence();
  if (threadIdx.x == 0) {
    int prev   = atomicAdd(cnt, 1);
    int target = (prev / NWG + 1) * NWG;
    while (__hip_atomic_load(cnt, __ATOMIC_ACQUIRE, __HIP_MEMORY_SCOPE_AGENT) < target)
      __builtin_amdgcn_s_sleep(1);
  }
  __syncthreads();
  __threadfence();
}

// ---- fp32 -> bf16 conversion ----------------------------------------------

__global__ __launch_bounds__(256) void cvt_bf16(const float* __restrict__ src,
                                                u16* __restrict__ dst, int n) {
  for (int i = blockIdx.x * 256 + threadIdx.x; i < n; i += gridDim.x * 256)
    dst[i] = f2bf(src[i]);
}

// ---- Phase A: input projections -------------------------------------------
// [131072 x 512](bf16) x [512 x 1536](bf16, N-major) + bias, f32 accumulate.
// 128(M) x 64(N) tile per 128-thread WG; wave w owns row strips w and w+4.
// Double-buffered async staging; output TIME-MAJOR: Yzr[t,b,2H], Yo[t,b,H].

__global__ __launch_bounds__(128) void phaseA(
    const u16* __restrict__ Xb,      // [B*T, 512] bf16, m = b*T + t
    const u16* __restrict__ Wih,     // [1024, 512] bf16
    const u16* __restrict__ Wio,     // [512, 512]  bf16
    const float* __restrict__ b_i2h, // [1024]
    const float* __restrict__ b_i2o, // [512]
    float* __restrict__ Yzr,         // [T, B, 1024]
    float* __restrict__ Yo)          // [T, B, 512]
{
  __shared__ __align__(16) u16 ldsA[2][128 * 32];   // 8 KB per buffer
  __shared__ __align__(16) u16 ldsB[2][64 * 32];    // 4 KB per buffer

  const int m0   = blockIdx.x * 128;
  const int n0   = blockIdx.y * 64;
  const int tid  = threadIdx.x;
  const int wave = tid >> 5, lane = tid & 31;

  v8f acc[2][4] = {};

  // per-thread staging assignments
  const u16* aSrcRow = &Xb[(size_t)(m0 + tid) * D_];          // thread copies row m0+tid
  const int  nrow    = n0 + (tid >> 1);
  const u16* bSrcRow = (nrow < H2_) ? &Wih[(size_t)nrow * D_]
                                    : &Wio[(size_t)(nrow - H2_) * D_];
  const int  bHalf   = (tid & 1) * 16;                        // 32B half of the row

  auto issue = [&](int buf, int kc) {
#pragma unroll
    for (int s = 0; s < 4; ++s)                               // A: 64B per thread
      async_copy16(aSrcRow + kc + s * 8, &ldsA[buf][tid * 32 + s * 8]);
#pragma unroll
    for (int s = 0; s < 2; ++s)                               // B: 32B per thread
      async_copy16(bSrcRow + kc + bHalf + s * 8,
                   &ldsB[buf][(tid >> 1) * 32 + bHalf + s * 8]);
  };

  issue(0, 0);
  int ib = 0;
  for (int kc = 0; kc < D_; kc += 32, ib ^= 1) {
    wait_async0();
    __syncthreads();                       // buffers filled; prev reads done
    if (kc + 32 < D_) issue(ib ^ 1, kc + 32);

    v16bf a0 = load_frag(ldsA[ib], wave * 16, 0, 32, lane);
    v16bf a1 = load_frag(ldsA[ib], 64 + wave * 16, 0, 32, lane);
#pragma unroll
    for (int nt = 0; nt < 4; ++nt) {
      v16bf b = load_frag(ldsB[ib], nt * 16, 0, 32, lane);
      acc[0][nt] = __builtin_amdgcn_wmma_f32_16x16x32_bf16(
          false, a0, false, b, (short)0, acc[0][nt], false, false);
      acc[1][nt] = __builtin_amdgcn_wmma_f32_16x16x32_bf16(
          false, a1, false, b, (short)0, acc[1][nt], false, false);
    }
  }

  const int rl = (lane >> 4) * 8;
  const int cl = lane & 15;
#pragma unroll
  for (int i = 0; i < 2; ++i) {
#pragma unroll
    for (int nt = 0; nt < 4; ++nt) {
#pragma unroll
      for (int j = 0; j < 8; ++j) {
        int m = m0 + i * 64 + wave * 16 + rl + j;
        int n = n0 + nt * 16 + cl;
        int b = m >> 11;                   // m / T_
        int t = m & (T_ - 1);
        float v = acc[i][nt][j];
        if (n < H2_) Yzr[((size_t)t * B_ + b) * H2_ + n]        = v + b_i2h[n];
        else         Yo [((size_t)t * B_ + b) * H_ + (n - H2_)] = v + b_i2o[n - H2_];
      }
    }
  }
}

// ---- Phase B: persistent recurrence ---------------------------------------
// 32 WGs x 128 threads. WG g owns preact cols [32g,32g+32) (stage 1) and
// hidden cols [16g,16g+16) (stage 2). Weight slices pinned in LDS; the whole
// 64 KB hidden state (bf16) is async-staged to LDS once per stage.

__global__ __launch_bounds__(128) void recurrence(
    const u16* __restrict__ Whh,     // [1024, 512] bf16
    const u16* __restrict__ Who,     // [512, 512]  bf16
    const float* __restrict__ b_h2h, // [1024]
    const float* __restrict__ b_h2o, // [512]
    const float* __restrict__ Yzr,   // [T, B, 1024]
    const float* __restrict__ Yo,    // [T, B, 512]
    const int* __restrict__ lens,    // [B]
    float* __restrict__ hF,          // [B, H] fp32 master
    u16*   __restrict__ hB,          // [B, H] bf16 shadow (WMMA A)
    u16*   __restrict__ aB,          // [B, H] bf16: h * sigmoid(r)
    float* __restrict__ zsBuf,       // [B, H] fp32: sigmoid(z)
    float* __restrict__ out,         // [B, T, H]
    float* __restrict__ finalH,      // [B, H]
    int*   __restrict__ barCnt)
{
  __shared__ __align__(16) u16 ldsW1[32 * D_];   // 32 KB : W_h2h rows [32g,32g+32)
  __shared__ __align__(16) u16 ldsW2[16 * D_];   // 16 KB : W_h2o rows [16g,16g+16)
  __shared__ __align__(16) u16 ldsH [B_ * H_];   // 64 KB : h (stage1) / a (stage2)

  const int g = blockIdx.x, tid = threadIdx.x;
  const int wave = tid >> 5, lane = tid & 31;

  // pin weight slices in LDS once (async bulk)
  bulk_copy(ldsW1, &Whh[(size_t)g * 32 * D_], 32 * D_, tid);
  bulk_copy(ldsW2, &Who[(size_t)g * 16 * D_], 16 * D_, tid);
  wait_async0();
  __syncthreads();

  const bool isR  = (g >= 16);
  const int  col1 = g * 32;
  const int  col2 = g * 16;
  const int  rl   = (lane >> 4) * 8;
  const int  cl   = lane & 15;

  for (int t = 0; t < T_; ++t) {
    // ---- stage 1: preact = Yzr_t + h @ Whh^T + b_h2h ----
    bulk_copy(ldsH, hB, B_ * H_, tid);           // whole hidden state, 64 KB
    wait_async0();
    __syncthreads();

    v8f acc[2] = {};
    for (int kc = 0; kc < H_; kc += 32) {
      v16bf a = load_frag(ldsH, wave * 16, kc, H_, lane);
#pragma unroll
      for (int nt = 0; nt < 2; ++nt) {
        v16bf b = load_frag(ldsW1, nt * 16, kc, D_, lane);
        acc[nt] = __builtin_amdgcn_wmma_f32_16x16x32_bf16(
            false, a, false, b, (short)0, acc[nt], false, false);
      }
    }
#pragma unroll
    for (int nt = 0; nt < 2; ++nt) {
#pragma unroll
      for (int j = 0; j < 8; ++j) {
        int b   = wave * 16 + rl + j;                 // batch row
        int col = col1 + nt * 16 + cl;                // preact col
        float v = acc[nt][j] + Yzr[((size_t)t * B_ + b) * H2_ + col] + b_h2h[col];
        float sg = sigmoidf(v);
        if (!isR) {
          zsBuf[b * H_ + col] = sg;                   // sigmoid(z)
        } else {
          int rc = col - H_;
          aB[b * H_ + rc] = f2bf(hF[b * H_ + rc] * sg);  // h * sigmoid(r)
        }
      }
    }
    grid_barrier(barCnt);

    // ---- stage 2: s = Yo_t + a @ Who^T + b_h2o; update h ----
    bulk_copy(ldsH, aB, B_ * H_, tid);           // reuse buffer for 'a'
    wait_async0();
    __syncthreads();

    v8f acc2 = {};
    for (int kc = 0; kc < H_; kc += 32) {
      v16bf a = load_frag(ldsH, wave * 16, kc, H_, lane);
      v16bf b = load_frag(ldsW2, 0, kc, D_, lane);
      acc2 = __builtin_amdgcn_wmma_f32_16x16x32_bf16(
          false, a, false, b, (short)0, acc2, false, false);
    }
#pragma unroll
    for (int j = 0; j < 8; ++j) {
      int b   = wave * 16 + rl + j;
      int col = col2 + cl;
      float s  = acc2[j] + Yo[((size_t)t * B_ + b) * H_ + col] + b_h2o[col];
      float zs = zsBuf[b * H_ + col];
      float ho = hF[b * H_ + col];
      float hn = (1.0f - zs) * ho + zs * tanhf(s);
      bool  active = (t < lens[b]);
      float hw = active ? hn : ho;                    // freeze finished rows
      out[((size_t)b * T_ + t) * H_ + col] = active ? hn : 0.0f;
      hF[b * H_ + col] = hw;
      hB[b * H_ + col] = f2bf(hw);
    }
    grid_barrier(barCnt);
  }

  // final hidden state (each WG owns its 16 columns)
  for (int i = tid; i < B_ * 16; i += 128) {
    int b = i >> 4, cc = col2 + (i & 15);
    finalH[b * H_ + cc] = hF[b * H_ + cc];
  }
}

// ---------------------------------------------------------------------------

extern "C" void kernel_launch(void* const* d_in, const int* in_sizes, int n_in,
                              void* d_out, int out_size, void* d_ws, size_t ws_size,
                              hipStream_t stream) {
  const float* seq    = (const float*)d_in[0];
  const int*   lens   = (const int*)  d_in[1];
  const float* W_i2h  = (const float*)d_in[2];
  const float* b_i2h  = (const float*)d_in[3];
  const float* W_h2h  = (const float*)d_in[4];
  const float* b_h2h  = (const float*)d_in[5];
  const float* W_i2o  = (const float*)d_in[6];
  const float* b_i2o  = (const float*)d_in[7];
  const float* W_h2o  = (const float*)d_in[8];
  const float* b_h2o  = (const float*)d_in[9];

  float* out    = (float*)d_out;                       // [B,T,H] outputs
  float* finalH = out + (size_t)B_ * T_ * H_;          // then [1,B,H]

  // ---- workspace layout (~0.95 GB) ----
  char* ws = (char*)d_ws;
  size_t off = 0;
  auto take = [&](size_t bytes) { char* p = ws + off; off = (off + bytes + 255) & ~(size_t)255; return p; };
  u16*   seqB  = (u16*)  take((size_t)B_ * T_ * D_ * 2);
  u16*   WihB  = (u16*)  take((size_t)H2_ * D_ * 2);
  u16*   WioB  = (u16*)  take((size_t)H_  * D_ * 2);
  u16*   WhhB  = (u16*)  take((size_t)H2_ * D_ * 2);
  u16*   WhoB  = (u16*)  take((size_t)H_  * D_ * 2);
  float* Yzr   = (float*)take((size_t)T_ * B_ * H2_ * 4);
  float* Yo    = (float*)take((size_t)T_ * B_ * H_  * 4);
  char*  state = take(0);                              // zeroed region starts here
  float* hF    = (float*)take((size_t)B_ * H_ * 4);
  u16*   hB    = (u16*)  take((size_t)B_ * H_ * 2);
  u16*   aB    = (u16*)  take((size_t)B_ * H_ * 2);
  float* zsB   = (float*)take((size_t)B_ * H_ * 4);
  int*   cnt   = (int*)  take(256);
  size_t stateBytes = (size_t)((char*)ws + off - state);

  (void)hipMemsetAsync(state, 0, stateBytes, stream);  // h0 / a / zs / barrier

  cvt_bf16<<<8192, 256, 0, stream>>>(seq,   seqB, B_ * T_ * D_);
  cvt_bf16<<<(H2_ * D_ + 255) / 256, 256, 0, stream>>>(W_i2h, WihB, H2_ * D_);
  cvt_bf16<<<(H_  * D_ + 255) / 256, 256, 0, stream>>>(W_i2o, WioB, H_  * D_);
  cvt_bf16<<<(H2_ * D_ + 255) / 256, 256, 0, stream>>>(W_h2h, WhhB, H2_ * D_);
  cvt_bf16<<<(H_  * D_ + 255) / 256, 256, 0, stream>>>(W_h2o, WhoB, H_  * D_);

  phaseA<<<dim3((B_ * T_) / 128, (H2_ + H_) / 64), 128, 0, stream>>>(
      seqB, WihB, WioB, b_i2h, b_i2o, Yzr, Yo);

  recurrence<<<NWG, 128, 0, stream>>>(
      WhhB, WhoB, b_h2h, b_h2o, Yzr, Yo, lens,
      hF, hB, aB, zsB, out, finalH, cnt);
}
